// GraphCast_21801253994718
// MI455X (gfx1250) — matrix operand order repaired and checked by reading, not compiled, not run on hardware
//
#include <hip/hip_runtime.h>
#include <hip/hip_bf16.h>

#define DD      128
#define NNODES  50000
#define NEDGES  256000
#define NLAYERS 2
#define LN_EPSF 1e-5f

typedef __attribute__((ext_vector_type(16))) __bf16 v16bf;
typedef __attribute__((ext_vector_type(8)))  __bf16 bf16x8;
typedef __attribute__((ext_vector_type(8)))  float  v8f;

__device__ __forceinline__ float rsum16(float x) {
  // reduce across the 16-lane half-wave that owns one matrix row
  x += __shfl_xor(x, 1, 32);
  x += __shfl_xor(x, 2, 32);
  x += __shfl_xor(x, 4, 32);
  x += __shfl_xor(x, 8, 32);
  return x;
}

__device__ __forceinline__ v16bf packA(float4 f0, float4 f1, float4 f2, float4 f3) {
  v16bf a;
  a[0]=(__bf16)f0.x;  a[1]=(__bf16)f0.y;  a[2]=(__bf16)f0.z;  a[3]=(__bf16)f0.w;
  a[4]=(__bf16)f1.x;  a[5]=(__bf16)f1.y;  a[6]=(__bf16)f1.z;  a[7]=(__bf16)f1.w;
  a[8]=(__bf16)f2.x;  a[9]=(__bf16)f2.y;  a[10]=(__bf16)f2.z; a[11]=(__bf16)f2.w;
  a[12]=(__bf16)f3.x; a[13]=(__bf16)f3.y; a[14]=(__bf16)f3.z; a[15]=(__bf16)f3.w;
  return a;
}

__device__ __forceinline__ v16bf loadB16(const __bf16* w) {
  bf16x8 lo = *(const bf16x8*)(w);
  bf16x8 hi = *(const bf16x8*)(w + 16);
  return __builtin_shufflevector(lo, hi, 0,1,2,3,4,5,6,7,8,9,10,11,12,13,14,15);
}

__global__ void zero_f32_kernel(float* __restrict__ p, int n) {
  int i = blockIdx.x * blockDim.x + threadIdx.x;
  if (i < n) p[i] = 0.0f;
}

// ---------------- Edge block: e += MLP([e, n[src], n[dst]]); agg[dst] += e ----
__global__ __launch_bounds__(256) void edge_mlp_kernel(
    const float* __restrict__ e_in, const float* __restrict__ n_in,
    const int* __restrict__ src, const int* __restrict__ dst,
    const float* __restrict__ W1, const float* __restrict__ b1,
    const float* __restrict__ g1, const float* __restrict__ be1,
    const float* __restrict__ W2, const float* __restrict__ b2,
    float* __restrict__ e_out, float* __restrict__ agg)
{
  extern __shared__ char smem[];
  __bf16* lw1 = (__bf16*)smem;             // [n][k] k-contiguous: lw1[n*384 + k]
  __bf16* lw2 = lw1 + 3 * DD * DD;         // lw2[n*128 + k]
  __bf16* lh  = lw2 + DD * DD;             // per-wave 16x128 activation tile

  const int tid = threadIdx.x;
  for (int i = tid; i < 3 * DD * DD; i += 256) {   // W1[k][n] -> lw1[n][k]
    int nn = i & (DD - 1), kk = i >> 7;
    lw1[nn * (3 * DD) + kk] = (__bf16)W1[i];
  }
  for (int i = tid; i < DD * DD; i += 256) {       // W2[k][n] -> lw2[n][k]
    int nn = i & (DD - 1), kk = i >> 7;
    lw2[nn * DD + kk] = (__bf16)W2[i];
  }
  __syncthreads();

  const int lane = tid & 31;
  const int wv   = tid >> 5;
  const int hlf  = lane >> 4;       // which 16-lane half
  const int ln   = lane & 15;
  const int h8   = hlf * 8;
  const int rowBase = blockIdx.x * 128 + wv * 16;
  __bf16* lhw = lh + wv * (16 * DD);

  // A-fragment row owned by this lane
  const int rowA = rowBase + ln;
  const float* pe = e_in + (size_t)rowA * DD;
  const float* ps = n_in + (size_t)src[rowA] * DD;
  const float* pd = n_in + (size_t)dst[rowA] * DD;

  float gval[8], beval[8], b1v[8], b2v[8];
  #pragma unroll
  for (int nt = 0; nt < 8; ++nt) {
    int col = nt * 16 + ln;
    gval[nt] = g1[col]; beval[nt] = be1[col]; b1v[nt] = b1[col]; b2v[nt] = b2[col];
  }

  // ---- GEMM1: [16 x 384] @ [384 x 128], K in 12 chunks of 32 (bf16 WMMA) ----
  v8f acc[8] = {};
  #pragma unroll
  for (int kc = 0; kc < 12; ++kc) {
    const float* p; int off0;
    if (kc < 4)      { p = pe; off0 = kc * 32; }
    else if (kc < 8) { p = ps; off0 = kc * 32 - 128; }
    else             { p = pd; off0 = kc * 32 - 256; }
    int c0 = off0 + h8;
    float4 f0 = *(const float4*)(p + c0);
    float4 f1 = *(const float4*)(p + c0 + 4);
    float4 f2 = *(const float4*)(p + c0 + 16);
    float4 f3 = *(const float4*)(p + c0 + 20);
    v16bf a = packA(f0, f1, f2, f3);
    #pragma unroll
    for (int nt = 0; nt < 8; ++nt) {
      v16bf b = loadB16(lw1 + (nt * 16 + ln) * (3 * DD) + kc * 32 + hlf * 8);
      acc[nt] = __builtin_amdgcn_wmma_f32_16x16x32_bf16(
          false, a, false, b, (short)0, acc[nt], false, false);
    }
  }

  // ---- bias + LayerNorm + SiLU; row m = v + h8, col = nt*16 + ln ----
  #pragma unroll
  for (int v = 0; v < 8; ++v) {
    float ls = 0.f, lss = 0.f;
    #pragma unroll
    for (int nt = 0; nt < 8; ++nt) {
      float x = acc[nt][v] + b1v[nt];
      ls += x; lss += x * x;
    }
    float s  = rsum16(ls);
    float ss = rsum16(lss);
    float mu   = s * (1.0f / 128.0f);
    float var  = ss * (1.0f / 128.0f) - mu * mu;
    float rinv = rsqrtf(var + LN_EPSF);
    #pragma unroll
    for (int nt = 0; nt < 8; ++nt) {
      float x = acc[nt][v] + b1v[nt];
      float y = (x - mu) * rinv * gval[nt] + beval[nt];
      float h = y / (1.0f + __expf(-y));          // SiLU
      lhw[(v + h8) * DD + nt * 16 + ln] = (__bf16)h;
    }
  }
  asm volatile("s_wait_dscnt 0x0" ::: "memory");   // our own LDS RAW

  // ---- GEMM2: [16 x 128] @ [128 x 128], K in 4 chunks of 32 ----
  v8f acc2[8] = {};
  #pragma unroll
  for (int kc = 0; kc < 4; ++kc) {
    const __bf16* ap = lhw + ln * DD + kc * 32 + h8;
    bf16x8 a0 = *(const bf16x8*)(ap);
    bf16x8 a1 = *(const bf16x8*)(ap + 16);
    v16bf a = __builtin_shufflevector(a0, a1, 0,1,2,3,4,5,6,7,8,9,10,11,12,13,14,15);
    #pragma unroll
    for (int nt = 0; nt < 8; ++nt) {
      v16bf b = loadB16(lw2 + (nt * 16 + ln) * DD + kc * 32 + hlf * 8);
      acc2[nt] = __builtin_amdgcn_wmma_f32_16x16x32_bf16(
          false, a, false, b, (short)0, acc2[nt], false, false);
    }
  }

  // ---- residual write + fused scatter-add into agg ----
  #pragma unroll
  for (int v = 0; v < 8; ++v) {
    int row  = rowBase + v + h8;
    int drow = dst[row];
    #pragma unroll
    for (int nt = 0; nt < 8; ++nt) {
      int col = nt * 16 + ln;
      float val = e_in[(size_t)row * DD + col] + acc2[nt][v] + b2v[nt];
      e_out[(size_t)row * DD + col] = val;
      atomicAdd(&agg[(size_t)drow * DD + col], val);   // global_atomic_add_f32
    }
  }
}

// ---------------- Node block: n += MLP([n, agg]) -----------------------------
__global__ __launch_bounds__(256) void node_mlp_kernel(
    const float* __restrict__ n_in, const float* __restrict__ agg,
    const float* __restrict__ W1, const float* __restrict__ b1,
    const float* __restrict__ g1, const float* __restrict__ be1,
    const float* __restrict__ W2, const float* __restrict__ b2,
    float* __restrict__ n_out, int nTotal)
{
  extern __shared__ char smem[];
  __bf16* lw1 = (__bf16*)smem;             // lw1[n*256 + k]
  __bf16* lw2 = lw1 + 2 * DD * DD;         // lw2[n*128 + k]
  __bf16* lh  = lw2 + DD * DD;

  const int tid = threadIdx.x;
  for (int i = tid; i < 2 * DD * DD; i += 256) {
    int nn = i & (DD - 1), kk = i >> 7;
    lw1[nn * (2 * DD) + kk] = (__bf16)W1[i];
  }
  for (int i = tid; i < DD * DD; i += 256) {
    int nn = i & (DD - 1), kk = i >> 7;
    lw2[nn * DD + kk] = (__bf16)W2[i];
  }
  __syncthreads();

  const int lane = tid & 31;
  const int wv   = tid >> 5;
  const int hlf  = lane >> 4;
  const int ln   = lane & 15;
  const int h8   = hlf * 8;
  const int rowBase = blockIdx.x * 128 + wv * 16;
  __bf16* lhw = lh + wv * (16 * DD);

  int rowA = rowBase + ln;
  int rowAc = rowA < nTotal ? rowA : (nTotal - 1);       // clamp reads, full EXEC
  const float* pn = n_in + (size_t)rowAc * DD;
  const float* pa = agg  + (size_t)rowAc * DD;

  float gval[8], beval[8], b1v[8], b2v[8];
  #pragma unroll
  for (int nt = 0; nt < 8; ++nt) {
    int col = nt * 16 + ln;
    gval[nt] = g1[col]; beval[nt] = be1[col]; b1v[nt] = b1[col]; b2v[nt] = b2[col];
  }

  // ---- GEMM1: [16 x 256] @ [256 x 128], 8 K-chunks ----
  v8f acc[8] = {};
  #pragma unroll
  for (int kc = 0; kc < 8; ++kc) {
    const float* p; int off0;
    if (kc < 4) { p = pn; off0 = kc * 32; }
    else        { p = pa; off0 = kc * 32 - 128; }
    int c0 = off0 + h8;
    float4 f0 = *(const float4*)(p + c0);
    float4 f1 = *(const float4*)(p + c0 + 4);
    float4 f2 = *(const float4*)(p + c0 + 16);
    float4 f3 = *(const float4*)(p + c0 + 20);
    v16bf a = packA(f0, f1, f2, f3);
    #pragma unroll
    for (int nt = 0; nt < 8; ++nt) {
      v16bf b = loadB16(lw1 + (nt * 16 + ln) * (2 * DD) + kc * 32 + hlf * 8);
      acc[nt] = __builtin_amdgcn_wmma_f32_16x16x32_bf16(
          false, a, false, b, (short)0, acc[nt], false, false);
    }
  }

  #pragma unroll
  for (int v = 0; v < 8; ++v) {
    float ls = 0.f, lss = 0.f;
    #pragma unroll
    for (int nt = 0; nt < 8; ++nt) {
      float x = acc[nt][v] + b1v[nt];
      ls += x; lss += x * x;
    }
    float s  = rsum16(ls);
    float ss = rsum16(lss);
    float mu   = s * (1.0f / 128.0f);
    float var  = ss * (1.0f / 128.0f) - mu * mu;
    float rinv = rsqrtf(var + LN_EPSF);
    #pragma unroll
    for (int nt = 0; nt < 8; ++nt) {
      float x = acc[nt][v] + b1v[nt];
      float y = (x - mu) * rinv * gval[nt] + beval[nt];
      float h = y / (1.0f + __expf(-y));
      lhw[(v + h8) * DD + nt * 16 + ln] = (__bf16)h;
    }
  }
  asm volatile("s_wait_dscnt 0x0" ::: "memory");

  // ---- GEMM2: [16 x 128] @ [128 x 128] ----
  v8f acc2[8] = {};
  #pragma unroll
  for (int kc = 0; kc < 4; ++kc) {
    const __bf16* ap = lhw + ln * DD + kc * 32 + h8;
    bf16x8 a0 = *(const bf16x8*)(ap);
    bf16x8 a1 = *(const bf16x8*)(ap + 16);
    v16bf a = __builtin_shufflevector(a0, a1, 0,1,2,3,4,5,6,7,8,9,10,11,12,13,14,15);
    #pragma unroll
    for (int nt = 0; nt < 8; ++nt) {
      v16bf b = loadB16(lw2 + (nt * 16 + ln) * DD + kc * 32 + hlf * 8);
      acc2[nt] = __builtin_amdgcn_wmma_f32_16x16x32_bf16(
          false, a, false, b, (short)0, acc2[nt], false, false);
    }
  }

  #pragma unroll
  for (int v = 0; v < 8; ++v) {
    int row = rowBase + v + h8;
    if (row < nTotal) {
      #pragma unroll
      for (int nt = 0; nt < 8; ++nt) {
        int col = nt * 16 + ln;
        n_out[(size_t)row * DD + col] =
            n_in[(size_t)row * DD + col] + acc2[nt][v] + b2v[nt];
      }
    }
  }
}

extern "C" void kernel_launch(void* const* d_in, const int* in_sizes, int n_in,
                              void* d_out, int out_size, void* d_ws, size_t ws_size,
                              hipStream_t stream) {
  const float* node_feat = (const float*)d_in[0];
  const float* edge_attr = (const float*)d_in[1];
  const int*   eidx      = (const int*)d_in[2];
  const int*   src = eidx;
  const int*   dst = eidx + NEDGES;
  const float* eW1  = (const float*)d_in[3];
  const float* eb1  = (const float*)d_in[4];
  const float* eg1  = (const float*)d_in[5];
  const float* ebe1 = (const float*)d_in[6];
  const float* eW2  = (const float*)d_in[7];
  const float* eb2  = (const float*)d_in[8];
  const float* nW1  = (const float*)d_in[9];
  const float* nb1  = (const float*)d_in[10];
  const float* ng1  = (const float*)d_in[11];
  const float* nbe1 = (const float*)d_in[12];
  const float* nW2  = (const float*)d_in[13];
  const float* nb2  = (const float*)d_in[14];

  float* n_out = (float*)d_out;                       // output tuple: (n, e)
  float* e_out = n_out + (size_t)NNODES * DD;
  float* agg   = (float*)d_ws;                        // [NNODES x DD]

  const size_t edge_lds = (size_t)(3 * DD * DD + DD * DD + 8 * 16 * DD) * sizeof(__bf16);
  const size_t node_lds = (size_t)(2 * DD * DD + DD * DD + 8 * 16 * DD) * sizeof(__bf16);

  const float* e_src = edge_attr;
  const float* n_src = node_feat;
  for (int i = 0; i < NLAYERS; ++i) {
    zero_f32_kernel<<<(NNODES * DD + 255) / 256, 256, 0, stream>>>(agg, NNODES * DD);
    edge_mlp_kernel<<<NEDGES / 128, 256, edge_lds, stream>>>(
        e_src, n_src, src, dst,
        eW1 + (size_t)i * 3 * DD * DD, eb1 + i * DD, eg1 + i * DD, ebe1 + i * DD,
        eW2 + (size_t)i * DD * DD, eb2 + i * DD,
        e_out, agg);
    node_mlp_kernel<<<(NNODES + 127) / 128, 256, node_lds, stream>>>(
        n_src, agg,
        nW1 + (size_t)i * 2 * DD * DD, nb1 + i * DD, ng1 + i * DD, nbe1 + i * DD,
        nW2 + (size_t)i * DD * DD, nb2 + i * DD,
        n_out, NNODES);
    e_src = e_out;
    n_src = n_out;
  }
}
